// UltimateHybridSuperLayer_48808008351928
// MI455X (gfx1250) — compile-verified, matrix operand order checked by reading, not run on hardware
//
#include <hip/hip_runtime.h>
#include <math.h>

// ---------------------------------------------------------------------------
// Types for CDNA5 WMMA (wave32, 16x16x32 bf16 -> f32)
// ---------------------------------------------------------------------------
typedef __attribute__((ext_vector_type(16))) __bf16 v16bf;
typedef __attribute__((ext_vector_type(8)))  __bf16 v8bf;
typedef __attribute__((ext_vector_type(8)))  float  v8f;

union FragBF16 { v16bf v; v8bf h[2]; };

__device__ __forceinline__ unsigned short f32_to_bf16(float f) {
    union { float f; unsigned int u; } cvt; cvt.f = f;
    unsigned int u = cvt.u;
    u += 0x7FFFu + ((u >> 16) & 1u);   // round-to-nearest-even
    return (unsigned short)(u >> 16);
}

__device__ __forceinline__ float gelu_exact(float x) {
    return 0.5f * x * (1.0f + erff(x * 0.70710678118654752f));
}

// ---------------------------------------------------------------------------
// gfx1250 async global->LDS (ASYNCcnt-tracked) via inline asm.
// VGLOBAL async encoding: VDST = LDS byte-offset VGPR, VADDR = 64-bit address.
// (ROCm 7.2's builtin has a non-portable typed-pointer signature; asm bypasses
// it, as recommended for the TDM builtins in the bridge doc.)
// ---------------------------------------------------------------------------
__device__ __forceinline__ void async_g2l_b128(unsigned short* lds_dst,
                                               const unsigned short* gsrc) {
    unsigned int loff = (unsigned int)(unsigned long long)
        ((__attribute__((address_space(3))) unsigned short*)lds_dst);
    asm volatile("global_load_async_to_lds_b128 %0, %1, off"
                 :: "v"(loff), "v"(gsrc) : "memory");
}

template <int N>
__device__ __forceinline__ void wait_asynccnt() {
#if __has_builtin(__builtin_amdgcn_s_wait_asynccnt)
    __builtin_amdgcn_s_wait_asynccnt(N);
#else
    asm volatile("s_wait_asynccnt %0" :: "i"(N) : "memory");
#endif
}

// ---------------------------------------------------------------------------
// Pack kernels (f32 -> bf16 staging, with zero padding)
// ---------------------------------------------------------------------------
__global__ void pack_pad_bf16(const float* __restrict__ src,
                              unsigned short* __restrict__ dst,
                              int R, int C, int Cpad) {
    size_t i = (size_t)blockIdx.x * blockDim.x + threadIdx.x;
    size_t total = (size_t)R * Cpad;
    if (i >= total) return;
    int r = (int)(i / Cpad);
    int c = (int)(i % Cpad);
    dst[i] = (c < C) ? f32_to_bf16(src[(size_t)r * C + c]) : (unsigned short)0;
}

// dst[n, k] = bf16(U[k, n] * S[n]) for n < R else 0   (dst is [Npad, K])
__global__ void pack_scaledT_bf16(const float* __restrict__ U,  // [K, R]
                                  const float* __restrict__ S,  // [R]
                                  unsigned short* __restrict__ dst, // [Npad, K]
                                  int K, int R, int Npad) {
    size_t i = (size_t)blockIdx.x * blockDim.x + threadIdx.x;
    size_t total = (size_t)Npad * K;
    if (i >= total) return;
    int n = (int)(i / K);
    int k = (int)(i % K);
    float v = (n < R) ? U[(size_t)k * R + n] * S[n] : 0.0f;
    dst[i] = f32_to_bf16(v);
}

// ---------------------------------------------------------------------------
// NT GEMM: C[M,N] = A[M,K] @ Bt[N,K]^T, bf16 in, f32 accumulate via WMMA.
//   EPI 0: store bf16 (no bias)
//   EPI 1: store bf16( gelu(acc + bias[n]) )
//   EPI 2: store f32 ( acc + bias[n] )
// BM=128 fixed; BN in {128, 256}. 256 threads = 8 waves (wave32).
//   BN=128: waves 4(M)x2(N), wave tile 32x64  (2x4 frags)
//   BN=256: waves 2(M)x4(N), wave tile 64x64  (4x4 frags)
// Global->LDS via async DMA loads, 3-stage LDS ring, 1 barrier per K-step.
// M % BM == 0, N % BN == 0, K % 32 == 0 guaranteed by padding.
// ---------------------------------------------------------------------------
template <int EPI, int BN>
__global__ void __launch_bounds__(256)
gemm_nt_bf16(const unsigned short* __restrict__ A,
             const unsigned short* __restrict__ Bt,
             void* __restrict__ Cout,
             const float* __restrict__ bias,
             int M, int N, int K) {
    constexpr int BM    = 128;
    constexpr int TK    = 32;
    constexpr int WCOLS = (BN == 256) ? 4 : 2;    // waves along N
    constexpr int WROWS = 8 / WCOLS;              // waves along M
    constexpr int MFRAG = BM / (WROWS * 16);      // 16-row frags per wave (M)
    constexpr int NFRAG = BN / (WCOLS * 16);      // 16-col frags per wave (N)
    constexpr int AP    = BM / 64;                // async b128 per thread (A)
    constexpr int BP    = BN / 64;                // async b128 per thread (B)
    constexpr int G     = AP + BP;                // async ops per tile group

    __shared__ __align__(16) unsigned short As[3][BM * TK];
    __shared__ __align__(16) unsigned short Bs[3][BN * TK];

    const int tid  = threadIdx.x;
    const int lane = tid & 31;
    const int wave = tid >> 5;
    const int wm   = wave / WCOLS;
    const int wn   = wave % WCOLS;
    const int r    = lane & 15;     // row/col within fragment
    const int half = lane >> 4;     // 0: K{0..7,16..23}  1: K{8..15,24..31}
    const int kb   = half * 8;

    const int blockN = blockIdx.x * BN;
    const int blockM = blockIdx.y * BM;

    // async tile loading: thread covers rows lr0 + p*64, 16B chunk lc
    const int lr0 = tid >> 2;        // 0..63
    const int lc  = (tid & 3) * 8;   // 0,8,16,24 (bf16 elems)

    const unsigned short* Aptr = A  + ((size_t)blockM + lr0) * K + lc;
    const unsigned short* Bptr = Bt + ((size_t)blockN + lr0) * K + lc;
    const size_t rstep = (size_t)64 * K;

    auto issue_tile = [&](int kk, int sb) {
#pragma unroll
        for (int p = 0; p < AP; ++p)
            async_g2l_b128(&As[sb][(lr0 + p * 64) * TK + lc], Aptr + kk + p * rstep);
#pragma unroll
        for (int p = 0; p < BP; ++p)
            async_g2l_b128(&Bs[sb][(lr0 + p * 64) * TK + lc], Bptr + kk + p * rstep);
    };

    v8f acc[MFRAG][NFRAG];
#pragma unroll
    for (int i = 0; i < MFRAG; ++i)
#pragma unroll
        for (int j = 0; j < NFRAG; ++j) acc[i][j] = v8f{};

    // prologue: start the DMA pipeline two tiles deep
    issue_tile(0, 0);
    if (TK < K) issue_tile(TK, 1);

    int s = 0;
    for (int k0 = 0; k0 < K; k0 += TK) {
        // Ensure this wave's tile-i DMA group completed (in-order retirement:
        // with tiles i and i+1 outstanding, asynccnt <= G proves tile i done).
        if (k0 + TK < K) wait_asynccnt<G>();
        else             wait_asynccnt<0>();
        __syncthreads();   // all waves' portions landed; old buffer reads done

        if (k0 + 2 * TK < K) {
            int s2 = s + 2; if (s2 >= 3) s2 -= 3;
            issue_tile(k0 + 2 * TK, s2);   // DMA overlaps WMMA below
        }

        // compute from stage s (ISA 16-bit A/B fragment layout)
        FragBF16 fa[MFRAG];
#pragma unroll
        for (int mi = 0; mi < MFRAG; ++mi) {
            const int row = wm * (MFRAG * 16) + mi * 16 + r;
            fa[mi].h[0] = *(const v8bf*)(&As[s][row * TK + kb]);
            fa[mi].h[1] = *(const v8bf*)(&As[s][row * TK + kb + 16]);
        }
#pragma unroll
        for (int ni = 0; ni < NFRAG; ++ni) {
            FragBF16 fb;
            const int row = wn * (NFRAG * 16) + ni * 16 + r;
            fb.h[0] = *(const v8bf*)(&Bs[s][row * TK + kb]);
            fb.h[1] = *(const v8bf*)(&Bs[s][row * TK + kb + 16]);
#pragma unroll
            for (int mi = 0; mi < MFRAG; ++mi)
                acc[mi][ni] = __builtin_amdgcn_wmma_f32_16x16x32_bf16(
                    /*neg_a=*/false, fa[mi].v, /*neg_b=*/false, fb.v,
                    /*c_mod=*/(short)0, acc[mi][ni],
                    /*reuse_a=*/false, /*reuse_b=*/false);
        }

        s = (s == 2) ? 0 : s + 1;
    }

    // Epilogue: C layout — VGPR j: lanes 0-15 -> M=j, lanes 16-31 -> M=j+8
    const int gm0 = blockM + wm * (MFRAG * 16) + half * 8;
    const int gn0 = blockN + wn * (NFRAG * 16) + r;
#pragma unroll
    for (int mi = 0; mi < MFRAG; ++mi) {
#pragma unroll
        for (int ni = 0; ni < NFRAG; ++ni) {
            const int col = gn0 + ni * 16;
            float bv = (EPI >= 1) ? bias[col] : 0.0f;
#pragma unroll
            for (int j = 0; j < 8; ++j) {
                const int row = gm0 + mi * 16 + j;
                float v = acc[mi][ni][j];
                if (EPI >= 1) v += bv;
                if (EPI == 1) v = gelu_exact(v);
                if (EPI == 2) {
                    ((float*)Cout)[(size_t)row * N + col] = v;
                } else {
                    ((unsigned short*)Cout)[(size_t)row * N + col] = f32_to_bf16(v);
                }
            }
        }
    }
}

// ---------------------------------------------------------------------------
// Host orchestration
// ---------------------------------------------------------------------------
extern "C" void kernel_launch(void* const* d_in, const int* in_sizes, int n_in,
                              void* d_out, int out_size, void* d_ws, size_t ws_size,
                              hipStream_t stream) {
    const float* x      = (const float*)d_in[0];
    const float* U_fc   = (const float*)d_in[1];
    const float* S_fc   = (const float*)d_in[2];
    const float* V_fc   = (const float*)d_in[3];
    const float* b_fc   = (const float*)d_in[4];
    const float* U_proj = (const float*)d_in[5];
    const float* S_proj = (const float*)d_in[6];
    const float* V_proj = (const float*)d_in[7];
    const float* b_proj = (const float*)d_in[8];

    constexpr int Mtot = 2 * 4096;   // B*S rows
    constexpr int DM   = 2048;       // d_model
    constexpr int DF   = 8192;       // d_ff
    constexpr int RK   = 819;        // rank
    constexpr int RP   = 896;        // rank padded to multiple of 128

    unsigned char* wsp = (unsigned char*)d_ws;
    auto carve = [&](size_t bytes) -> unsigned short* {
        unsigned short* p = (unsigned short*)wsp;
        wsp += (bytes + 255) & ~(size_t)255;
        return p;
    };
    unsigned short* xb   = carve((size_t)Mtot * DM * 2); // x bf16            [8192,2048]
    unsigned short* Bt1  = carve((size_t)RP   * DM * 2); // (U_fc*S)^T bf16   [896,2048]
    unsigned short* t1   = carve((size_t)Mtot * RP * 2); // x@(U S)           [8192,896]
    unsigned short* Vfcb = carve((size_t)DF   * RP * 2); // V_fc bf16 pad     [8192,896]
    unsigned short* hbuf = carve((size_t)Mtot * DF * 2); // gelu hidden bf16  [8192,8192]
    unsigned short* Bt3  = carve((size_t)RP   * DF * 2); // (U_proj*S)^T bf16 [896,8192]
    unsigned short* t2   = carve((size_t)Mtot * RP * 2); // h@(U S)           [8192,896]
    unsigned short* Vprb = carve((size_t)DM   * RP * 2); // V_proj bf16 pad   [2048,896]

    const int TPB = 256;
    auto blocks1d = [&](size_t n) { return dim3((unsigned)((n + TPB - 1) / TPB)); };

    // Stage 0: pack/convert all operands to bf16 with zero padding
    pack_pad_bf16<<<blocks1d((size_t)Mtot * DM), TPB, 0, stream>>>(x, xb, Mtot, DM, DM);
    pack_scaledT_bf16<<<blocks1d((size_t)RP * DM), TPB, 0, stream>>>(U_fc, S_fc, Bt1, DM, RK, RP);
    pack_pad_bf16<<<blocks1d((size_t)DF * RP), TPB, 0, stream>>>(V_fc, Vfcb, DF, RK, RP);
    pack_scaledT_bf16<<<blocks1d((size_t)RP * DF), TPB, 0, stream>>>(U_proj, S_proj, Bt3, DF, RK, RP);
    pack_pad_bf16<<<blocks1d((size_t)DM * RP), TPB, 0, stream>>>(V_proj, Vprb, DM, RK, RP);

    // Stage 1: t1 = x @ (U_fc * S_fc)            [8192, 896]  (bf16 out, BN=128)
    gemm_nt_bf16<0, 128><<<dim3(RP / 128, Mtot / 128), TPB, 0, stream>>>(
        xb, Bt1, t1, nullptr, Mtot, RP, DM);

    // Stage 2: h = gelu(t1 @ V_fc^T + b_fc)      [8192, 8192] (bf16 out, BN=256)
    gemm_nt_bf16<1, 256><<<dim3(DF / 256, Mtot / 128), TPB, 0, stream>>>(
        t1, Vfcb, hbuf, b_fc, Mtot, DF, RP);

    // Stage 3: t2 = h @ (U_proj * S_proj)        [8192, 896]  (bf16 out, BN=128)
    gemm_nt_bf16<0, 128><<<dim3(RP / 128, Mtot / 128), TPB, 0, stream>>>(
        hbuf, Bt3, t2, nullptr, Mtot, RP, DF);

    // Stage 4: out = t2 @ V_proj^T + b_proj      [8192, 2048] (f32 out, BN=256)
    gemm_nt_bf16<2, 256><<<dim3(DM / 256, Mtot / 128), TPB, 0, stream>>>(
        t2, Vprb, (float*)d_out, b_proj, Mtot, DM, RP);
}
